// MessagePassingAffinityModel_89833535963854
// MI455X (gfx1250) — matrix operand order repaired and checked by reading, not compiled, not run on hardware
//
#include <hip/hip_runtime.h>

// ---------------- problem constants (match reference) ----------------
#define N_NODES 65536
#define N_EDGES 1048576
#define NB      64
#define HD      128
#define NL      3
#define K1      257       // 2H+1
#define K1P     288       // padded to multiple of 32
#define KU      256       // 2H

typedef __attribute__((ext_vector_type(16))) _Float16 v16h;
typedef __attribute__((ext_vector_type(8)))  float    v8f;

// ---- WMMA fragment helpers (wave32, 16x16x32 f16) ----
// A layout (ISA 7.12.2, 16-bit A 16x32): lanes 0-15 and 16-31 both hold rows
// M=0..15; VGPR j<4 holds K = 2j + 8*half, VGPR j>=4 holds K = 16 + 2(j-4) + 8*half.
__device__ __forceinline__ v16h load_a_frag(const _Float16* rowPtr, int k0, int half) {
  union { v16h v; unsigned int u[8]; } t;
  const unsigned int* p = (const unsigned int*)rowPtr;
  const int b = (k0 >> 1) + half * 4;
#pragma unroll
  for (int j = 0; j < 4; ++j) t.u[j] = p[b + j];
#pragma unroll
  for (int j = 0; j < 4; ++j) t.u[4 + j] = p[b + 8 + j];
  return t.v;
}

// B layout (matches A): VGPR j<4 holds K = 2j + 8*half, VGPR j>=4 holds
// K = 16 + 2(j-4) + 8*half, for column N=lane%16 (column-major weights).
__device__ __forceinline__ v16h load_b_frag(const _Float16* colPtr, int k0, int half) {
  union { v16h v; unsigned int u[8]; } t;
  const unsigned int* p = (const unsigned int*)colPtr + (k0 >> 1) + half * 4;
#pragma unroll
  for (int j = 0; j < 4; ++j) t.u[j] = p[j];
#pragma unroll
  for (int j = 0; j < 4; ++j) t.u[4 + j] = p[8 + j];
  return t.v;
}

__device__ __forceinline__ v8f wmma16(v16h a, v16h b, v8f c) {
  return __builtin_amdgcn_wmma_f32_16x16x32_f16(false, a, false, b, (short)0, c,
                                                false, false);
}

// Packed f16 atomic add (CDNA5 global_atomic_pk_add_f16): 2 channels per RMW,
// halving the segment-sum atomic op count and traffic vs f32 atomics.
__device__ __forceinline__ void atomic_pk_add_f16(_Float16* addr, unsigned int packed) {
  asm volatile("global_atomic_pk_add_f16 %0, %1, off"
               :: "v"(addr), "v"(packed)
               : "memory");
}

// CDNA5 async memory->LDS copy (GLOBAL_LOAD_ASYNC_TO_LDS_B128, ASYNCcnt):
// 16 bytes per lane, no VGPR staging. lds_addr = per-lane LDS byte address
// (low 32 bits of the generic pointer), gaddr = 64-bit global address.
__device__ __forceinline__ void async_copy16(unsigned int lds_addr, const void* gaddr) {
  asm volatile("global_load_async_to_lds_b128 %0, %1, off"
               :: "v"(lds_addr), "v"(gaddr)
               : "memory");
}
__device__ __forceinline__ void wait_async0() {
  asm volatile("s_wait_asynccnt 0x0" ::: "memory");
}
__device__ __forceinline__ unsigned int lds_addr_of(const void* p) {
  return (unsigned int)(unsigned long long)p;   // LDS byte address is the low 32 bits of the pointer
}

// ---------------- small utility kernels ----------------
__global__ void zero_f32_kernel(float* __restrict__ p, int n) {
  int i = blockIdx.x * blockDim.x + threadIdx.x;
  if (i < n) p[i] = 0.f;
}

// transpose + f16-convert + K-pad:  src [K, Nn] f32 -> dst [Nn, Kpad] f16
__global__ void convert_wT_kernel(const float* __restrict__ src, _Float16* __restrict__ dst,
                                  int K, int Nn, int Kpad) {
  int i = blockIdx.x * blockDim.x + threadIdx.x;
  if (i >= Nn * Kpad) return;
  int nn = i / Kpad, k = i - nn * Kpad;
  dst[i] = (k < K) ? (_Float16)src[k * Nn + nn] : (_Float16)0.f;
}

__global__ void centroid_accum_kernel(const float* __restrict__ pos, const int* __restrict__ batch,
                                      float* __restrict__ csum, float* __restrict__ ccnt) {
  int i = blockIdx.x * blockDim.x + threadIdx.x;
  if (i >= N_NODES) return;
  int b = batch[i];
  atomicAdd(&csum[b * 3 + 0], pos[i * 3 + 0]);
  atomicAdd(&csum[b * 3 + 1], pos[i * 3 + 1]);
  atomicAdd(&csum[b * 3 + 2], pos[i * 3 + 2]);
  atomicAdd(&ccnt[b], 1.f);
}

__global__ void centroid_fin_kernel(const float* __restrict__ csum, const float* __restrict__ ccnt,
                                    float* __restrict__ center) {
  int i = threadIdx.x;
  if (i < NB * 3) center[i] = csum[i] / fmaxf(ccnt[i / 3], 1.f);
}

__global__ void posrel_kernel(const float* __restrict__ pos, const int* __restrict__ batch,
                              const float* __restrict__ center, float* __restrict__ pos_rel) {
  int i = blockIdx.x * blockDim.x + threadIdx.x;
  if (i >= N_NODES * 3) return;
  int n = i / 3, d = i - n * 3;
  pos_rel[i] = pos[i] - center[batch[n] * 3 + d];
}

// x0 = [emb[z], pos_rel] @ lin_in_w + lin_in_b  -> f16   (2.2 GFLOP; not hot)
__global__ void lin_in_kernel(const float* __restrict__ emb, const int* __restrict__ z,
                              const float* __restrict__ pos_rel, const float* __restrict__ W,
                              const float* __restrict__ bias, _Float16* __restrict__ xh) {
  int i = blockIdx.x * blockDim.x + threadIdx.x;   // N*H threads; 128 thr share a node
  int n = i >> 7, h = i & 127;
  const float* er = emb + (size_t)z[n] * HD;
  float s = bias[h];
#pragma unroll 4
  for (int k = 0; k < HD; ++k) s = fmaf(er[k], W[k * HD + h], s);
  const float* pr = pos_rel + n * 3;
#pragma unroll
  for (int d = 0; d < 3; ++d) s = fmaf(pr[d], W[(HD + d) * HD + h], s);
  xh[i] = (_Float16)s;
}

__global__ void edge_geom_kernel(const float* __restrict__ pos_rel, const int* __restrict__ row,
                                 const int* __restrict__ col, float* __restrict__ dist,
                                 float* __restrict__ deg) {
  int i = blockIdx.x * blockDim.x + threadIdx.x;
  if (i >= N_EDGES) return;
  int r = row[i], c = col[i];
  float dx = pos_rel[c * 3 + 0] - pos_rel[r * 3 + 0];
  float dy = pos_rel[c * 3 + 1] - pos_rel[r * 3 + 1];
  float dz = pos_rel[c * 3 + 2] - pos_rel[r * 3 + 2];
  dist[i] = sqrtf(dx * dx + dy * dy + dz * dz);
  atomicAdd(&deg[r], 1.f);
}

// ---------------- fused edge MLP (WMMA) ----------------
// 128 edges per 128-thread block (4 waves, 32 rows / wave = 2 M-tiles).
// W1/W2 preloaded into LDS once per block (CDNA5: 320 KB LDS per WGP);
// gather + preload use async global->LDS DMA (no VGPR staging);
// segment-sum scatter uses packed-f16 global atomics (2 ch / RMW).
__global__ __launch_bounds__(128) void edge_mlp_kernel(
    const _Float16* __restrict__ xh, const float* __restrict__ dist,
    const int* __restrict__ row, const int* __restrict__ col,
    const _Float16* __restrict__ W1t, const float* __restrict__ b1,
    const _Float16* __restrict__ W2t, const float* __restrict__ b2,
    _Float16* __restrict__ msum)
{
  constexpr int MT    = 128;  // edges per block
  constexpr int ASTR  = 296;  // K1P=288 + pad (16B-aligned rows, breaks bank stride)
  constexpr int WSTR1 = 296;
  constexpr int HSTR  = 136;  // HD=128 + pad
  constexpr int WSTR2 = 136;
  __shared__ _Float16 sA[MT * ASTR];      // 75,776 B
  __shared__ _Float16 sW1[HD * WSTR1];    // 75,776 B
  __shared__ _Float16 sH[MT * HSTR];      // 34,816 B (H1, later reused for M2)
  __shared__ _Float16 sW2[HD * WSTR2];    // 34,816 B
  __shared__ int sRow[MT];                //    512 B  -> ~221 KB total

  const int t  = threadIdx.x;
  const int e0 = blockIdx.x * MT;

  // Phase 0: weight preload via async DMA (one column per thread)
  {
    const unsigned int lw1 = lds_addr_of(sW1 + t * WSTR1);
    const _Float16* g1 = W1t + (size_t)t * K1P;
#pragma unroll
    for (int i = 0; i < (K1P * 2) / 16; ++i) async_copy16(lw1 + i * 16, g1 + i * 8);
    const unsigned int lw2 = lds_addr_of(sW2 + t * WSTR2);
    const _Float16* g2 = W2t + (size_t)t * HD;
#pragma unroll
    for (int i = 0; i < (HD * 2) / 16; ++i) async_copy16(lw2 + i * 16, g2 + i * 8);
  }
  // Phase 1: gather [x[col] | x[row] | dist | 0-pad] via async DMA (1 edge/thread)
  {
    const int e = e0 + t;
    const int rs = row[e], cs = col[e];
    sRow[t] = rs;
    const unsigned int la = lds_addr_of(sA + t * ASTR);
    const _Float16* gc = xh + (size_t)cs * HD;
    const _Float16* gr = xh + (size_t)rs * HD;
#pragma unroll
    for (int i = 0; i < 16; ++i) {
      async_copy16(la + i * 16, gc + i * 8);
      async_copy16(la + 256 + i * 16, gr + i * 8);
    }
    sA[t * ASTR + 2 * HD] = (_Float16)dist[e];        // disjoint LDS bytes
#pragma unroll
    for (int k = 2 * HD + 1; k < K1P; ++k) sA[t * ASTR + k] = (_Float16)0.f;
  }
  wait_async0();          // async fills complete (ASYNCcnt == 0)
  __syncthreads();        // visible to all waves

  const int wave = t >> 5, lane = t & 31;
  const int half = lane >> 4, idx = lane & 15;
  const int r0 = wave * 32;                         // 32-row band per wave
  const _Float16* aRow0 = sA + (r0 + idx) * ASTR;
  const _Float16* aRow1 = sA + (r0 + 16 + idx) * ASTR;

  // Phase 2: H1 = relu(A @ W1 + b1)   (K=288, 9 k-steps, 8 N-tiles, 2 M-tiles)
  v8f acc0[8] = {}, acc1[8] = {};
  for (int k0 = 0; k0 < K1P; k0 += 32) {
    v16h a0 = load_a_frag(aRow0, k0, half);
    v16h a1 = load_a_frag(aRow1, k0, half);
#pragma unroll
    for (int nt = 0; nt < 8; ++nt) {
      v16h b = load_b_frag(sW1 + (nt * 16 + idx) * WSTR1, k0, half);
      acc0[nt] = wmma16(a0, b, acc0[nt]);           // B reused by both M-tiles
      acc1[nt] = wmma16(a1, b, acc1[nt]);
    }
  }
#pragma unroll
  for (int nt = 0; nt < 8; ++nt) {
    const int n = nt * 16 + idx;
    const float bv = b1[n];
#pragma unroll
    for (int r = 0; r < 8; ++r) {                   // D: M = r + half*8, N = idx
      sH[(r0 + r + half * 8) * HSTR + n]      = (_Float16)fmaxf(acc0[nt][r] + bv, 0.f);
      sH[(r0 + 16 + r + half * 8) * HSTR + n] = (_Float16)fmaxf(acc1[nt][r] + bv, 0.f);
    }
  }
  __syncthreads();

  // Phase 3: M2 = relu(H1 @ W2 + b2)
  const _Float16* hRow0 = sH + (r0 + idx) * HSTR;
  const _Float16* hRow1 = sH + (r0 + 16 + idx) * HSTR;
  v8f m0[8] = {}, m1[8] = {};
  for (int k0 = 0; k0 < HD; k0 += 32) {
    v16h a0 = load_a_frag(hRow0, k0, half);
    v16h a1 = load_a_frag(hRow1, k0, half);
#pragma unroll
    for (int nt = 0; nt < 8; ++nt) {
      v16h b = load_b_frag(sW2 + (nt * 16 + idx) * WSTR2, k0, half);
      m0[nt] = wmma16(a0, b, m0[nt]);
      m1[nt] = wmma16(a1, b, m1[nt]);
    }
  }
  // Stage relu'd messages back into sH (rows are wave-local; DS ops of a wave
  // are in-order, and this wave's k-loop reads of sH are already complete).
#pragma unroll
  for (int nt = 0; nt < 8; ++nt) {
    const int n = nt * 16 + idx;
    const float bv = b2[n];
#pragma unroll
    for (int r = 0; r < 8; ++r) {
      sH[(r0 + r + half * 8) * HSTR + n]      = (_Float16)fmaxf(m0[nt][r] + bv, 0.f);
      sH[(r0 + 16 + r + half * 8) * HSTR + n] = (_Float16)fmaxf(m1[nt][r] + bv, 0.f);
    }
  }
  // Phase 4: packed-f16 atomic scatter; thread t owns edge row t (row t is in
  // wave (t>>5)'s band, so no cross-wave LDS dependency).
  {
    const int node = sRow[t];
    const unsigned int* hrow = (const unsigned int*)(sH + t * HSTR);
    _Float16* dst = msum + (size_t)node * HD;
#pragma unroll 4
    for (int c = 0; c < HD / 2; ++c) {
      atomic_pk_add_f16(dst + 2 * c, hrow[c]);
    }
  }
}

// ---------------- node update (WMMA): x = relu([x | msum/deg] @ Wu + bu) ----------------
__global__ __launch_bounds__(128) void node_update_kernel(
    _Float16* __restrict__ xh, const _Float16* __restrict__ msum,
    const float* __restrict__ deg, const _Float16* __restrict__ Wut,
    const float* __restrict__ bu)
{
  constexpr int ASTR = 264;   // KU=256 + pad
  __shared__ _Float16 sA[64 * ASTR];
  const int t = threadIdx.x;
  const int n0 = blockIdx.x * 64;
  {
    const int ln = t >> 1, which = t & 1;
    const int node = n0 + ln;
    if (which == 0) {
      const uint4* s = (const uint4*)(xh + (size_t)node * HD);
      uint4* d = (uint4*)(sA + ln * ASTR);
#pragma unroll
      for (int i = 0; i < 16; ++i) d[i] = s[i];
    } else {
      const float invd = 1.f / fmaxf(deg[node], 1.f);
      const _Float16* mr = msum + (size_t)node * HD;
#pragma unroll 4
      for (int k = 0; k < HD; ++k) sA[ln * ASTR + HD + k] = (_Float16)((float)mr[k] * invd);
    }
  }
  __syncthreads();

  const int wave = t >> 5, lane = t & 31;
  const int half = lane >> 4, idx = lane & 15;
  const _Float16* aRow = sA + (wave * 16 + idx) * ASTR;

  v8f acc[8] = {};
  for (int k0 = 0; k0 < KU; k0 += 32) {
    v16h a = load_a_frag(aRow, k0, half);
#pragma unroll
    for (int nt = 0; nt < 8; ++nt) {
      v16h b = load_b_frag(Wut + (size_t)(nt * 16 + idx) * KU, k0, half);
      acc[nt] = wmma16(a, b, acc[nt]);
    }
  }
#pragma unroll
  for (int nt = 0; nt < 8; ++nt) {
    const int n = nt * 16 + idx;
    const float bv = bu[n];
#pragma unroll
    for (int r = 0; r < 8; ++r) {
      const int node = n0 + wave * 16 + r + half * 8;
      xh[(size_t)node * HD + n] = (_Float16)fmaxf(acc[nt][r] + bv, 0.f);   // in-place, node-owned
    }
  }
}

// ---------------- readout ----------------
__global__ void lig_accum_kernel(const _Float16* __restrict__ xh, const int* __restrict__ ntype,
                                 const int* __restrict__ batch, float* __restrict__ gsum,
                                 float* __restrict__ gcnt) {
  int i = blockIdx.x * blockDim.x + threadIdx.x;   // N*H
  int n = i >> 7, h = i & 127;
  if (ntype[n] != 1) return;
  atomicAdd(&gsum[batch[n] * HD + h], (float)xh[i]);
  if (h == 0) atomicAdd(&gcnt[batch[n]], 1.f);
}

__global__ void readout1_kernel(const float* __restrict__ gsum, const float* __restrict__ gcnt,
                                const float* __restrict__ W, const float* __restrict__ bias,
                                float* __restrict__ hg) {
  int i = blockIdx.x * blockDim.x + threadIdx.x;   // B*H
  if (i >= NB * HD) return;
  int b = i >> 7, h = i & 127;
  float invc = 1.f / fmaxf(gcnt[b], 1.f);
  float s = bias[h];
#pragma unroll 4
  for (int k = 0; k < HD; ++k) s = fmaf(gsum[b * HD + k] * invc, W[k * HD + h], s);
  hg[i] = fmaxf(s, 0.f);
}

__global__ void readout2_kernel(const float* __restrict__ hg, const float* __restrict__ w2,
                                const float* __restrict__ b2, float* __restrict__ out) {
  __shared__ float red[HD];
  int b = blockIdx.x, t = threadIdx.x;
  red[t] = hg[b * HD + t] * w2[t];
  __syncthreads();
  for (int s = HD / 2; s > 0; s >>= 1) {
    if (t < s) red[t] += red[t + s];
    __syncthreads();
  }
  if (t == 0) out[b] = red[0] + b2[0];
}

// ---------------- host-side orchestration ----------------
extern "C" void kernel_launch(void* const* d_in, const int* in_sizes, int n_in,
                              void* d_out, int out_size, void* d_ws, size_t ws_size,
                              hipStream_t stream) {
  const float* pos      = (const float*)d_in[0];
  const int*   z        = (const int*)d_in[1];
  const int*   batch    = (const int*)d_in[2];
  const int*   ei       = (const int*)d_in[3];
  const int*   ntype    = (const int*)d_in[4];
  const float* emb      = (const float*)d_in[5];
  const float* lin_in_w = (const float*)d_in[6];
  const float* lin_in_b = (const float*)d_in[7];
  const float* msg_w1   = (const float*)d_in[8];
  const float* msg_b1   = (const float*)d_in[9];
  const float* msg_w2   = (const float*)d_in[10];
  const float* msg_b2   = (const float*)d_in[11];
  const float* upd_w    = (const float*)d_in[12];
  const float* upd_b    = (const float*)d_in[13];
  const float* ro_w1    = (const float*)d_in[14];
  const float* ro_b1    = (const float*)d_in[15];
  const float* ro_w2    = (const float*)d_in[16];
  const float* ro_b2    = (const float*)d_in[17];
  float* out = (float*)d_out;

  const int* row = ei;              // edge_index[0]
  const int* col = ei + N_EDGES;    // edge_index[1]

  char* base = (char*)d_ws;
  size_t off = 0;
  auto carve = [&](size_t bytes) -> void* {
    void* p = base + off;
    off = (off + bytes + 255) & ~(size_t)255;
    return p;
  };
  float*    csum    = (float*)carve(NB * 3 * 4);
  float*    ccnt    = (float*)carve(NB * 4);
  float*    center  = (float*)carve(NB * 3 * 4);
  float*    pos_rel = (float*)carve((size_t)N_NODES * 3 * 4);
  _Float16* xh      = (_Float16*)carve((size_t)N_NODES * HD * 2);
  float*    dist    = (float*)carve((size_t)N_EDGES * 4);
  float*    deg     = (float*)carve((size_t)N_NODES * 4);
  _Float16* msum    = (_Float16*)carve((size_t)N_NODES * HD * 2);
  _Float16* W1t     = (_Float16*)carve((size_t)NL * HD * K1P * 2);
  _Float16* W2t     = (_Float16*)carve((size_t)NL * HD * HD * 2);
  _Float16* Wut     = (_Float16*)carve((size_t)NL * HD * KU * 2);
  float*    gsum    = (float*)carve(NB * HD * 4);
  float*    gcnt    = (float*)carve(NB * 4);
  float*    hg      = (float*)carve(NB * HD * 4);

  const int NH = N_NODES * HD;

  // zero all accumulation buffers (ws is poisoned by the harness)
  zero_f32_kernel<<<1, 256, 0, stream>>>(csum, NB * 3);
  zero_f32_kernel<<<1, 256, 0, stream>>>(ccnt, NB);
  zero_f32_kernel<<<(N_NODES + 255) / 256, 256, 0, stream>>>(deg, N_NODES);
  zero_f32_kernel<<<(NB * HD + 255) / 256, 256, 0, stream>>>(gsum, NB * HD);
  zero_f32_kernel<<<1, 256, 0, stream>>>(gcnt, NB);

  // weight transpose/convert (f32 [K,N] -> f16 [N,Kpad])
  for (int l = 0; l < NL; ++l) {
    convert_wT_kernel<<<(HD * K1P + 255) / 256, 256, 0, stream>>>(
        msg_w1 + (size_t)l * K1 * HD, W1t + (size_t)l * HD * K1P, K1, HD, K1P);
    convert_wT_kernel<<<(HD * HD + 255) / 256, 256, 0, stream>>>(
        msg_w2 + (size_t)l * HD * HD, W2t + (size_t)l * HD * HD, HD, HD, HD);
    convert_wT_kernel<<<(HD * KU + 255) / 256, 256, 0, stream>>>(
        upd_w + (size_t)l * KU * HD, Wut + (size_t)l * HD * KU, KU, HD, KU);
  }

  // geometry + input embedding
  centroid_accum_kernel<<<N_NODES / 256, 256, 0, stream>>>(pos, batch, csum, ccnt);
  centroid_fin_kernel<<<1, 256, 0, stream>>>(csum, ccnt, center);
  posrel_kernel<<<(N_NODES * 3 + 255) / 256, 256, 0, stream>>>(pos, batch, center, pos_rel);
  lin_in_kernel<<<NH / 256, 256, 0, stream>>>(emb, z, pos_rel, lin_in_w, lin_in_b, xh);
  edge_geom_kernel<<<N_EDGES / 256, 256, 0, stream>>>(pos_rel, row, col, dist, deg);

  // message-passing layers
  for (int l = 0; l < NL; ++l) {
    zero_f32_kernel<<<(NH / 2 + 255) / 256, 256, 0, stream>>>((float*)msum, NH / 2);  // f16 zeros
    edge_mlp_kernel<<<N_EDGES / 128, 128, 0, stream>>>(
        xh, dist, row, col,
        W1t + (size_t)l * HD * K1P, msg_b1 + (size_t)l * HD,
        W2t + (size_t)l * HD * HD,  msg_b2 + (size_t)l * HD, msum);
    node_update_kernel<<<N_NODES / 64, 128, 0, stream>>>(
        xh, msum, deg, Wut + (size_t)l * HD * KU, upd_b + (size_t)l * HD);
  }

  // readout
  lig_accum_kernel<<<NH / 256, 256, 0, stream>>>(xh, ntype, batch, gsum, gcnt);
  readout1_kernel<<<(NB * HD + 255) / 256, 256, 0, stream>>>(gsum, gcnt, ro_w1, ro_b1, hg);
  readout2_kernel<<<NB, HD, 0, stream>>>(hg, ro_w2, ro_b2, out);
}
